// Attention_74543452389795
// MI455X (gfx1250) — matrix operand order repaired and checked by reading, not compiled, not run on hardware
//
#include <hip/hip_runtime.h>

// ---------------------------------------------------------------------------
// MHA forward for MI455X (gfx1250, wave32, WMMA bf16 16x16x32, f32 accum)
// ---------------------------------------------------------------------------

typedef __attribute__((ext_vector_type(16))) __bf16 v16bf;
typedef __attribute__((ext_vector_type(8)))  __bf16 v8bf;
typedef __attribute__((ext_vector_type(8)))  float  v8f;

#define DIM       1024
#define NUM_HEADS 16
#define HEAD_DIM  64
#define SEQ_N     2048
#define BATCH     4
#define NTOK      (BATCH * SEQ_N)        // 8192 tokens
#define SCALE     0.125f                 // HEAD_DIM^-0.5

// ---- WMMA fragment helpers (layouts per CDNA5 ISA §7.12.2) -----

// A fragment, 16x32 bf16 (M x K): lane holds row M = lane%16.
// Packed elems 0..7  -> K = k0 + (lane/16)*8 + 0..7
// Packed elems 8..15 -> K = k0 + (lane/16)*8 + 16..23
// => two contiguous 16-byte loads per lane.
__device__ __forceinline__ v16bf load_A(const __bf16* base, int ld,
                                        int row0, int k0, int lane) {
    const __bf16* p = base + (size_t)(row0 + (lane & 15)) * ld
                           + k0 + ((lane >> 4) << 3);
    union { v16bf v; v8bf h[2]; } u;
    u.h[0] = *(const v8bf*)(p);
    u.h[1] = *(const v8bf*)(p + 16);
    return u.v;
}

// B fragment, 32x16 bf16 (K x N): lane holds column N = lane%16.
// Packed elems 0..7  -> K = k0 + (lane/16)*8 + 0..7
// Packed elems 8..15 -> K = k0 + (lane/16)*8 + 16..23
__device__ __forceinline__ v16bf load_B(const __bf16* base, int ld,
                                        int k0, int col0, int lane) {
    const __bf16* p = base + (size_t)(k0 + ((lane >> 4) << 3)) * ld
                           + col0 + (lane & 15);
    v16bf r;
    #pragma unroll
    for (int s = 0; s < 8; ++s) {
        r[s]     = p[(size_t)s * ld];
        r[s + 8] = p[(size_t)(s + 16) * ld];
    }
    return r;
}

__device__ __forceinline__ v8f wmma_bf16(v16bf a, v16bf b, v8f c) {
    // (neg_a, A, neg_b, B, c_mod, C, reuse_a, reuse_b)
    return __builtin_amdgcn_wmma_f32_16x16x32_bf16(
        false, a, false, b, (short)0, c, false, false);
}

// C/D fragment mapping: elem r of lane l -> row = r + (l>>4)*8, col = l&15.

// ---------------------------------------------------------------------------
// Kernel 1a: f32 -> bf16 elementwise (x)
// ---------------------------------------------------------------------------
__global__ void k_cvt(const float* __restrict__ src, __bf16* __restrict__ dst,
                      int n) {
    int i = blockIdx.x * blockDim.x + threadIdx.x;
    if (i < n) dst[i] = (__bf16)src[i];
}

// Kernel 1b: W[o][i] f32 -> Wt[i][o] bf16 (so WMMA B frags are contiguous)
__global__ void k_cvt_wT(const float* __restrict__ W, __bf16* __restrict__ Wt) {
    int idx = blockIdx.x * blockDim.x + threadIdx.x;   // DIM*DIM threads
    int o = idx >> 10;
    int i = idx & (DIM - 1);
    Wt[(size_t)i * DIM + o] = (__bf16)W[idx];
}

// ---------------------------------------------------------------------------
// Kernel 2: QKV projection. One wave -> 16x64 output tile of one of q/k/v.
//   q -> qb[b][h][n][d]  (pre-scaled by SCALE)
//   k -> kt[b][h][d][n]  (transposed: B-frags for Q.K^T are contiguous)
//   v -> vb[b][h][n][d]
// ---------------------------------------------------------------------------
__global__ void k_qkv(const __bf16* __restrict__ xb,
                      const __bf16* __restrict__ wq,
                      const __bf16* __restrict__ wk,
                      const __bf16* __restrict__ wv,
                      __bf16* __restrict__ qb,
                      __bf16* __restrict__ kt,
                      __bf16* __restrict__ vb) {
    const int lane = threadIdx.x;
    const int tilesPerMat = (NTOK / 16) * (DIM / 64);   // 512 * 16 = 8192
    int bx  = blockIdx.x;
    int mat = bx / tilesPerMat;                         // 0=q 1=k 2=v
    int rem = bx - mat * tilesPerMat;
    int mt  = rem >> 4;                                 // token tile (16 rows)
    int nt  = rem & 15;                                 // head index (64 cols)
    int row0 = mt * 16;

    const __bf16* Wt = (mat == 0) ? wq : (mat == 1) ? wk : wv;

    v8f acc[4] = {};
    for (int k0 = 0; k0 < DIM; k0 += 32) {
        v16bf a = load_A(xb, DIM, row0, k0, lane);
        #pragma unroll
        for (int j = 0; j < 4; ++j) {
            v16bf b = load_B(Wt, DIM, k0, nt * 64 + 16 * j, lane);
            acc[j] = wmma_bf16(a, b, acc[j]);
        }
    }

    const int row_off = (lane >> 4) * 8;
    const int col     = lane & 15;
    #pragma unroll
    for (int j = 0; j < 4; ++j) {
        #pragma unroll
        for (int r = 0; r < 8; ++r) {
            int t  = row0 + row_off + r;
            int bb = t >> 11;                // /2048
            int n  = t & (SEQ_N - 1);
            int d  = 16 * j + col;
            float v = acc[j][r];
            if (mat == 0) {
                qb[(((size_t)bb * NUM_HEADS + nt) * SEQ_N + n) * HEAD_DIM + d] =
                    (__bf16)(v * SCALE);
            } else if (mat == 1) {
                kt[(((size_t)bb * NUM_HEADS + nt) * HEAD_DIM + d) * SEQ_N + n] =
                    (__bf16)v;
            } else {
                vb[(((size_t)bb * NUM_HEADS + nt) * SEQ_N + n) * HEAD_DIM + d] =
                    (__bf16)v;
            }
        }
    }
}

// ---------------------------------------------------------------------------
// Kernel 3: flash attention. One wave per 16-row Q block of one (b,h).
//   S = Q Kt (bf16 WMMA, f32 acc), online softmax, O += P V.
//   Writes O (softmax-normalized) as bf16 into ob[b][n][h*64+d].
// ---------------------------------------------------------------------------
__global__ void k_attn(const __bf16* __restrict__ qb,
                       const __bf16* __restrict__ kt,
                       const __bf16* __restrict__ vb,
                       __bf16* __restrict__ ob) {
    __shared__ __attribute__((aligned(32))) __bf16 lds_p[16 * 32];

    const int lane = threadIdx.x;
    const int qblocks = SEQ_N / 16;                      // 128
    int bh   = blockIdx.x / qblocks;                     // 0..63
    int qblk = blockIdx.x - bh * qblocks;
    int b = bh / NUM_HEADS;
    int h = bh - b * NUM_HEADS;

    const __bf16* Q  = qb + (size_t)bh * SEQ_N * HEAD_DIM + (size_t)qblk * 16 * HEAD_DIM;
    const __bf16* Kt = kt + (size_t)bh * HEAD_DIM * SEQ_N;
    const __bf16* V  = vb + (size_t)bh * SEQ_N * HEAD_DIM;

    // Q fragments for K=0..31 and K=32..63 (D=64 split into two WMMA k-steps)
    v16bf aq0 = load_A(Q, HEAD_DIM, 0, 0,  lane);
    v16bf aq1 = load_A(Q, HEAD_DIM, 0, 32, lane);

    v8f o0 = {}, o1 = {}, o2 = {}, o3 = {};
    float rm[8], rl[8];
    #pragma unroll
    for (int r = 0; r < 8; ++r) { rm[r] = -__builtin_inff(); rl[r] = 0.0f; }

    const int row_off = (lane >> 4) * 8;
    const int col     = lane & 15;

    for (int m0 = 0; m0 < SEQ_N; m0 += 32) {
        // ---- S = Q K^T for 16x32 KV chunk (two 16x16 tiles) ----
        v8f s0 = {}, s1 = {};
        {
            v16bf b00 = load_B(Kt, SEQ_N, 0,  m0,      lane);
            v16bf b10 = load_B(Kt, SEQ_N, 32, m0,      lane);
            v16bf b01 = load_B(Kt, SEQ_N, 0,  m0 + 16, lane);
            v16bf b11 = load_B(Kt, SEQ_N, 32, m0 + 16, lane);
            s0 = wmma_bf16(aq0, b00, s0);
            s0 = wmma_bf16(aq1, b10, s0);
            s1 = wmma_bf16(aq0, b01, s1);
            s1 = wmma_bf16(aq1, b11, s1);
        }

        // ---- online softmax (row stats across the 16 lanes of each row) ----
        float alpha[8];
        #pragma unroll
        for (int r = 0; r < 8; ++r) {
            float mx = fmaxf(s0[r], s1[r]);
            #pragma unroll
            for (int d = 1; d < 16; d <<= 1)
                mx = fmaxf(mx, __shfl_xor(mx, d, 32));
            float mn = fmaxf(rm[r], mx);
            alpha[r] = __expf(rm[r] - mn);
            float p0 = __expf(s0[r] - mn);
            float p1 = __expf(s1[r] - mn);
            s0[r] = p0; s1[r] = p1;
            float rs = p0 + p1;
            #pragma unroll
            for (int d = 1; d < 16; d <<= 1)
                rs += __shfl_xor(rs, d, 32);
            rl[r] = rl[r] * alpha[r] + rs;
            rm[r] = mn;
        }
        #pragma unroll
        for (int r = 0; r < 8; ++r) {
            o0[r] *= alpha[r]; o1[r] *= alpha[r];
            o2[r] *= alpha[r]; o3[r] *= alpha[r];
        }

        // ---- P: C-layout -> A-layout via LDS (bf16, row-major 16x32) ----
        #pragma unroll
        for (int r = 0; r < 8; ++r) {
            lds_p[(row_off + r) * 32 + col]      = (__bf16)s0[r];
            lds_p[(row_off + r) * 32 + col + 16] = (__bf16)s1[r];
        }
        __syncthreads();
        v16bf ap = load_A(lds_p, 32, 0, 0, lane);
        __syncthreads();

        // ---- O += P V (16x32 @ 32x64) ----
        const __bf16* Vm = V + (size_t)m0 * HEAD_DIM;
        o0 = wmma_bf16(ap, load_B(Vm, HEAD_DIM, 0, 0,  lane), o0);
        o1 = wmma_bf16(ap, load_B(Vm, HEAD_DIM, 0, 16, lane), o1);
        o2 = wmma_bf16(ap, load_B(Vm, HEAD_DIM, 0, 32, lane), o2);
        o3 = wmma_bf16(ap, load_B(Vm, HEAD_DIM, 0, 48, lane), o3);
    }

    // ---- normalize and store into concat layout ob[b][n][h*64+d] ----
    #pragma unroll
    for (int r = 0; r < 8; ++r) {
        float inv = 1.0f / rl[r];
        int n = qblk * 16 + row_off + r;
        size_t base = ((size_t)b * SEQ_N + n) * DIM + h * HEAD_DIM;
        ob[base +  0 + col] = (__bf16)(o0[r] * inv);
        ob[base + 16 + col] = (__bf16)(o1[r] * inv);
        ob[base + 32 + col] = (__bf16)(o2[r] * inv);
        ob[base + 48 + col] = (__bf16)(o3[r] * inv);
    }
}

// ---------------------------------------------------------------------------
// Kernel 4: output projection out = O @ Wo^T + bo (f32 output)
// ---------------------------------------------------------------------------
__global__ void k_out(const __bf16* __restrict__ ob,
                      const __bf16* __restrict__ wo,   // transposed [k][o]
                      const float*  __restrict__ bo,
                      float* __restrict__ out) {
    const int lane = threadIdx.x;
    int mt = blockIdx.x >> 4;
    int nt = blockIdx.x & 15;
    int row0 = mt * 16;

    v8f acc[4] = {};
    for (int k0 = 0; k0 < DIM; k0 += 32) {
        v16bf a = load_A(ob, DIM, row0, k0, lane);
        #pragma unroll
        for (int j = 0; j < 4; ++j) {
            v16bf b = load_B(wo, DIM, k0, nt * 64 + 16 * j, lane);
            acc[j] = wmma_bf16(a, b, acc[j]);
        }
    }

    const int row_off = (lane >> 4) * 8;
    const int col     = lane & 15;
    #pragma unroll
    for (int j = 0; j < 4; ++j) {
        #pragma unroll
        for (int r = 0; r < 8; ++r) {
            int t = row0 + row_off + r;
            int c = nt * 64 + 16 * j + col;
            out[(size_t)t * DIM + c] = acc[j][r] + bo[c];
        }
    }
}

// ---------------------------------------------------------------------------
extern "C" void kernel_launch(void* const* d_in, const int* in_sizes, int n_in,
                              void* d_out, int out_size, void* d_ws, size_t ws_size,
                              hipStream_t stream) {
    const float* x  = (const float*)d_in[0];
    const float* Wq = (const float*)d_in[1];
    const float* Wk = (const float*)d_in[2];
    const float* Wv = (const float*)d_in[3];
    const float* Wo = (const float*)d_in[4];
    const float* bo = (const float*)d_in[5];
    float* out = (float*)d_out;

    // workspace layout (bf16 elements)
    const size_t NX = (size_t)NTOK * DIM;        // 8M
    const size_t NW = (size_t)DIM * DIM;         // 1M
    const size_t NH = (size_t)BATCH * NUM_HEADS * SEQ_N * HEAD_DIM; // 8M
    __bf16* p   = (__bf16*)d_ws;
    __bf16* xb  = p;  p += NX;
    __bf16* wqT = p;  p += NW;
    __bf16* wkT = p;  p += NW;
    __bf16* wvT = p;  p += NW;
    __bf16* woT = p;  p += NW;
    __bf16* qb  = p;  p += NH;
    __bf16* kt  = p;  p += NH;
    __bf16* vb  = p;  p += NH;
    __bf16* ob  = xb;                            // x dead after QKV: alias

    // 1) precision conversion + weight transpose
    k_cvt<<<(int)((NX + 255) / 256), 256, 0, stream>>>(x, xb, (int)NX);
    k_cvt_wT<<<(int)(NW / 256), 256, 0, stream>>>(Wq, wqT);
    k_cvt_wT<<<(int)(NW / 256), 256, 0, stream>>>(Wk, wkT);
    k_cvt_wT<<<(int)(NW / 256), 256, 0, stream>>>(Wv, wvT);
    k_cvt_wT<<<(int)(NW / 256), 256, 0, stream>>>(Wo, woT);

    // 2) QKV projection (3 matrices * 512 row-tiles * 16 head-tiles)
    k_qkv<<<3 * (NTOK / 16) * (DIM / 64), 32, 0, stream>>>(
        xb, wqT, wkT, wvT, qb, kt, vb);

    // 3) flash attention: B*H*(N/16) = 8192 waves
    k_attn<<<BATCH * NUM_HEADS * (SEQ_N / 16), 32, 0, stream>>>(qb, kt, vb, ob);

    // 4) output projection + bias
    k_out<<<(NTOK / 16) * (DIM / 64), 32, 0, stream>>>(ob, woT, bo, out);
}
